// PointNetFeaturePropagation_52304111730781
// MI455X (gfx1250) — matrix operand order repaired and checked by reading, not compiled, not run on hardware
//
#include <hip/hip_runtime.h>
#include <stdint.h>

#define B_   16
#define N1_  4096
#define N2_  1024
#define C1_  128
#define C2_  256
#define CIN_ 384
#define M0_  256
#define M1_  128

typedef __attribute__((ext_vector_type(16))) __bf16 v16bf;
typedef __attribute__((ext_vector_type(8)))  float  v8f;

__device__ __forceinline__ unsigned short f32_to_bf16(float f) {
  unsigned int u = __float_as_uint(f);
  unsigned int r = u + 0x7FFFu + ((u >> 16) & 1u);   // round-to-nearest-even
  return (unsigned short)(r >> 16);
}
__device__ __forceinline__ float bf16_to_f32(unsigned short h) {
  return __uint_as_float(((unsigned int)h) << 16);
}

// ---------------------------------------------------------------- split fp32 -> bf16 hi/lo
__global__ void split_f32_kernel(const float* __restrict__ src,
                                 unsigned short* __restrict__ hi,
                                 unsigned short* __restrict__ lo, int n) {
  int i = blockIdx.x * 256 + threadIdx.x;
  if (i >= n) return;
  float v = src[i];
  unsigned short h = f32_to_bf16(v);
  hi[i] = h;
  lo[i] = f32_to_bf16(v - bf16_to_f32(h));
}

// ---------------------------------------------------------------- 3-NN (squared dist) per point
__global__ void knn3_kernel(const float* __restrict__ xyz1, const float* __restrict__ xyz2,
                            int* __restrict__ idx3, float* __restrict__ w3) {
  __shared__ float s[N2_ * 3];
  int b = blockIdx.y;
  int i = blockIdx.x * 256 + threadIdx.x;
  const float* x2 = xyz2 + (size_t)b * N2_ * 3;
  for (int t = threadIdx.x; t < N2_ * 3; t += 256) s[t] = x2[t];
  __syncthreads();
  const float* p = xyz1 + ((size_t)b * N1_ + i) * 3;
  float px = p[0], py = p[1], pz = p[2];
  float d0 = 1e30f, d1 = 1e30f, d2 = 1e30f;
  int   i0 = 0,     i1 = 0,     i2 = 0;
  for (int j = 0; j < N2_; ++j) {
    float dx = s[3 * j] - px, dy = s[3 * j + 1] - py, dz = s[3 * j + 2] - pz;
    float d = dx * dx + dy * dy + dz * dz;
    if (d < d2) {
      if (d < d1) {
        if (d < d0) { d2 = d1; i2 = i1; d1 = d0; i1 = i0; d0 = d; i0 = j; }
        else        { d2 = d1; i2 = i1; d1 = d;  i1 = j; }
      } else        { d2 = d;  i2 = j; }
    }
  }
  float w0 = 1.f / (d0 + 1e-8f), w1 = 1.f / (d1 + 1e-8f), w2 = 1.f / (d2 + 1e-8f);
  float inv = 1.f / (w0 + w1 + w2);
  size_t base = ((size_t)b * N1_ + i) * 3;
  idx3[base] = i0; idx3[base + 1] = i1; idx3[base + 2] = i2;
  w3[base] = w0 * inv; w3[base + 1] = w1 * inv; w3[base + 2] = w2 * inv;
}

// ------------------------------------------- gather/interp + concat + bf16 hi/lo split of X
__global__ void interp_concat_kernel(const float* __restrict__ feat1, const float* __restrict__ feat2,
                                     const int* __restrict__ idx3, const float* __restrict__ w3,
                                     unsigned short* __restrict__ Xhi, unsigned short* __restrict__ Xlo) {
  __shared__ int   sidx[128 * 3];
  __shared__ float sw[128 * 3];
  int b = blockIdx.y;
  int it0 = blockIdx.x * 128;
  size_t pbase = ((size_t)b * N1_ + it0) * 3;
  for (int t = threadIdx.x; t < 128 * 3; t += 256) { sidx[t] = idx3[pbase + t]; sw[t] = w3[pbase + t]; }
  __syncthreads();
  int lp = threadIdx.x & 127;
  int c0 = threadIdx.x >> 7;              // 0 or 1
  int i = it0 + lp;
  for (int c = c0; c < CIN_; c += 2) {
    float v;
    if (c < C1_) {
      v = feat1[((size_t)b * C1_ + c) * N1_ + i];
    } else {
      const float* f2 = feat2 + ((size_t)b * C2_ + (c - C1_)) * N2_;
      v = sw[lp * 3]     * f2[sidx[lp * 3]]
        + sw[lp * 3 + 1] * f2[sidx[lp * 3 + 1]]
        + sw[lp * 3 + 2] * f2[sidx[lp * 3 + 2]];
    }
    unsigned short h = f32_to_bf16(v);
    size_t o = ((size_t)b * CIN_ + c) * N1_ + i;
    Xhi[o] = h;
    Xlo[o] = f32_to_bf16(v - bf16_to_f32(h));
  }
}

// ---------------------------------------------------------------- WMMA GEMM (bf16x3 split)
#define TBM 128
#define TBN 64
#define TBK 32
#define RPAD 40      // padded LDS row stride (ushorts): 80B = 20 banks -> conflict-free frags, 16B aligned

// B-tile register prefetch: only 8 VGPRs live across the WMMA block
struct PrefB { uint4 h, l; };

__device__ __forceinline__ void b_global_load(PrefB& p,
    const unsigned short* __restrict__ XHb, const unsigned short* __restrict__ XLb,
    int n0, int bcol, int bks, int k0) {
  size_t gb = (size_t)(k0 + bks) * N1_ + n0 + bcol;
  union { uint4 q; unsigned short s[8]; } uh, ul;
#pragma unroll
  for (int t = 0; t < 8; ++t) {     // lanes coalesce: 32 consecutive ushorts per k-row
    uh.s[t] = XHb[gb + (size_t)t * N1_];
    ul.s[t] = XLb[gb + (size_t)t * N1_];
  }
  p.h = uh.q;                       // pack immediately; only 2 uint4 stay live
  p.l = ul.q;
}

__device__ __forceinline__ void b_stage(const PrefB& p,
    unsigned short* BsH, unsigned short* BsL, int bcol, int bks) {
  *(uint4*)&BsH[bcol * RPAD + bks] = p.h;   // single aligned b128 store (transposed tile)
  *(uint4*)&BsL[bcol * RPAD + bks] = p.l;
}

// A (weights) is L2-resident: load global -> store LDS directly at staging time
__device__ __forceinline__ void a_stage(
    const unsigned short* __restrict__ Ahi, const unsigned short* __restrict__ Alo,
    unsigned short* AsH, unsigned short* AsL,
    int K, int m0, int arow, int akk, int k0) {
  const uint4* pH = (const uint4*)(Ahi + (size_t)(m0 + arow) * K + k0 + akk);
  const uint4* pL = (const uint4*)(Alo + (size_t)(m0 + arow) * K + k0 + akk);
  uint4 a0 = pH[0], a1 = pH[1], a2 = pL[0], a3 = pL[1];
  *(uint4*)&AsH[arow * RPAD + akk]     = a0;
  *(uint4*)&AsH[arow * RPAD + akk + 8] = a1;
  *(uint4*)&AsL[arow * RPAD + akk]     = a2;
  *(uint4*)&AsL[arow * RPAD + akk + 8] = a3;
}

__device__ __forceinline__ v16bf load_a_frag(const unsigned short* As, int mbase, int hf, int r) {
  union { unsigned int u[8]; v16bf v; } f;
  const unsigned short* row = As + (mbase + r) * RPAD;
#pragma unroll
  for (int v = 0; v < 8; ++v) {
    int kk = ((v >> 2) << 4) + (hf << 3) + ((v & 3) << 1);   // K pair index
    f.u[v] = *(const unsigned int*)(row + kk);
  }
  return f.v;
}
__device__ __forceinline__ v16bf load_b_frag(const unsigned short* Bs, int nbase, int hf, int r) {
  union { unsigned int u[8]; v16bf v; } f;
  const unsigned short* row = Bs + (nbase + r) * RPAD;
#pragma unroll
  for (int v = 0; v < 8; ++v) f.u[v] = *(const unsigned int*)(row + ((v >> 2) << 4) + (hf << 3) + ((v & 3) << 1));
  return f.v;
}

__global__ __launch_bounds__(256)
__attribute__((amdgpu_waves_per_eu(2, 4)))   // relaxed VGPR budget
void wmma_gemm_kernel(const unsigned short* __restrict__ Ahi, const unsigned short* __restrict__ Alo,
                      const unsigned short* __restrict__ Xhi, const unsigned short* __restrict__ Xlo,
                      float* __restrict__ Y, int Mtot, int K) {
  __shared__ unsigned short AsH[2 * TBM * RPAD], AsL[2 * TBM * RPAD];
  __shared__ unsigned short BsH[2 * TBN * RPAD], BsL[2 * TBN * RPAD];
  int tid = threadIdx.x;
  int col0 = blockIdx.x * TBN;              // column over B*4096
  int b = col0 >> 12, n0 = col0 & (N1_ - 1);
  int m0 = blockIdx.y * TBM;
  const unsigned short* XHb = Xhi + (size_t)b * K * N1_;
  const unsigned short* XLb = Xlo + (size_t)b * K * N1_;
  float* Yb = Y + (size_t)b * Mtot * N1_;

  int lane = tid & 31, wave = tid >> 5;
  int wm = wave & 3, wn = wave >> 2;        // wave tile: 32x32
  int hf = lane >> 4, r = lane & 15;

  v8f acc[2][2];
#pragma unroll
  for (int i = 0; i < 2; ++i)
#pragma unroll
    for (int j = 0; j < 2; ++j)
#pragma unroll
      for (int v = 0; v < 8; ++v) acc[i][j][v] = 0.0f;

  int arow = tid >> 1, akk = (tid & 1) * 16;   // A stage: 2 threads / row
  int bcol = tid & 63, bks = (tid >> 6) * 8;   // B stage: thread owns column n, 8-wide K-seg

  const int steps = K / TBK;
  PrefB pb;
  // prologue: stage step 0 into buffer 0
  b_global_load(pb, XHb, XLb, n0, bcol, bks, 0);
  a_stage(Ahi, Alo, AsH, AsL, K, m0, arow, akk, 0);
  b_stage(pb, BsH, BsL, bcol, bks);
  __syncthreads();

  for (int s = 0; s < steps; ++s) {
    int cur = s & 1;
    if (s + 1 < steps)   // B prefetch (HBM latency) hides under the WMMAs below
      b_global_load(pb, XHb, XLb, n0, bcol, bks, (s + 1) * TBK);

    const unsigned short* aH_ = AsH + cur * TBM * RPAD;
    const unsigned short* aL_ = AsL + cur * TBM * RPAD;
    const unsigned short* bH_ = BsH + cur * TBN * RPAD;
    const unsigned short* bL_ = BsL + cur * TBN * RPAD;

    v16bf aH[2], aL[2], bH[2], bL[2];
#pragma unroll
    for (int i = 0; i < 2; ++i) {
      aH[i] = load_a_frag(aH_, wm * 32 + i * 16, hf, r);
      aL[i] = load_a_frag(aL_, wm * 32 + i * 16, hf, r);
    }
#pragma unroll
    for (int j = 0; j < 2; ++j) {
      bH[j] = load_b_frag(bH_, wn * 32 + j * 16, hf, r);
      bL[j] = load_b_frag(bL_, wn * 32 + j * 16, hf, r);
    }
#pragma unroll
    for (int i = 0; i < 2; ++i)
#pragma unroll
      for (int j = 0; j < 2; ++j) {
        acc[i][j] = __builtin_amdgcn_wmma_f32_16x16x32_bf16(false, aL[i], false, bH[j], (short)0, acc[i][j], false, false);
        acc[i][j] = __builtin_amdgcn_wmma_f32_16x16x32_bf16(false, aH[i], false, bL[j], (short)0, acc[i][j], false, false);
        acc[i][j] = __builtin_amdgcn_wmma_f32_16x16x32_bf16(false, aH[i], false, bH[j], (short)0, acc[i][j], false, false);
      }

    if (s + 1 < steps) {
      int nxt = cur ^ 1;
      // A is L2-hot: direct global->LDS staging here; B comes from the register prefetch
      a_stage(Ahi, Alo, AsH + nxt * TBM * RPAD, AsL + nxt * TBM * RPAD, K, m0, arow, akk, (s + 1) * TBK);
      b_stage(pb, BsH + nxt * TBN * RPAD, BsL + nxt * TBN * RPAD, bcol, bks);
      __syncthreads();   // one barrier per K-step
    }
  }

#pragma unroll
  for (int i = 0; i < 2; ++i)
#pragma unroll
    for (int j = 0; j < 2; ++j)
#pragma unroll
      for (int v = 0; v < 8; ++v) {
        int m = m0 + wm * 32 + i * 16 + (hf << 3) + v;   // D layout: M = v + 8*(lane>=16)
        int n = n0 + wn * 32 + j * 16 + r;
        Yb[(size_t)m * N1_ + n] = acc[i][j][v];
      }
}

// ---------------------------------------------------------------- per-channel batch stats -> scale/shift
__global__ void bn_stats_kernel(const float* __restrict__ Y, const float* __restrict__ gamma,
                                const float* __restrict__ beta, float* __restrict__ scale,
                                float* __restrict__ shift, int M) {
  __shared__ float ssum[256], ssq[256];
  int m = blockIdx.x, tid = threadIdx.x;
  float s = 0.f, q = 0.f;
  for (int t = tid; t < B_ * N1_; t += 256) {
    int b = t >> 12, n = t & (N1_ - 1);
    float v = Y[((size_t)b * M + m) * N1_ + n];
    s += v; q += v * v;
  }
  ssum[tid] = s; ssq[tid] = q;
  __syncthreads();
  for (int st = 128; st > 0; st >>= 1) {
    if (tid < st) { ssum[tid] += ssum[tid + st]; ssq[tid] += ssq[tid + st]; }
    __syncthreads();
  }
  if (tid == 0) {
    const float invN = 1.f / (float)(B_ * N1_);
    float mean = ssum[0] * invN;
    float var = ssq[0] * invN - mean * mean;      // biased, matches jnp.var
    float sc = gamma[m] * rsqrtf(var + 1e-5f);
    scale[m] = sc;
    shift[m] = beta[m] - mean * sc;               // conv bias cancels under batch-stat BN
  }
}

// ---------------------------------------------------------------- BN+ReLU -> bf16 hi/lo (mid layer)
__global__ void bn_relu_split_kernel(const float* __restrict__ Y, const float* __restrict__ scale,
                                     const float* __restrict__ shift,
                                     unsigned short* __restrict__ Hhi, unsigned short* __restrict__ Hlo,
                                     int mmask) {
  size_t idx = (size_t)blockIdx.x * 256 + threadIdx.x;
  int m = (int)((idx >> 12) & mmask);
  float v = fmaxf(0.f, Y[idx] * scale[m] + shift[m]);
  unsigned short h = f32_to_bf16(v);
  Hhi[idx] = h;
  Hlo[idx] = f32_to_bf16(v - bf16_to_f32(h));
}

// ---------------------------------------------------------------- BN+ReLU -> fp32 output
__global__ void bn_relu_out_kernel(const float* __restrict__ Y, const float* __restrict__ scale,
                                   const float* __restrict__ shift, float* __restrict__ out, int mmask) {
  size_t idx = (size_t)blockIdx.x * 256 + threadIdx.x;
  int m = (int)((idx >> 12) & mmask);
  out[idx] = fmaxf(0.f, Y[idx] * scale[m] + shift[m]);
}

// ================================================================ host launcher
extern "C" void kernel_launch(void* const* d_in, const int* in_sizes, int n_in,
                              void* d_out, int out_size, void* d_ws, size_t ws_size,
                              hipStream_t stream) {
  const float* xyz1  = (const float*)d_in[0];
  const float* xyz2  = (const float*)d_in[1];
  const float* feat1 = (const float*)d_in[2];
  const float* feat2 = (const float*)d_in[3];
  const float* w0    = (const float*)d_in[4];
  const float* g0    = (const float*)d_in[6];
  const float* be0   = (const float*)d_in[7];
  const float* w1    = (const float*)d_in[8];
  const float* g1    = (const float*)d_in[10];
  const float* be1   = (const float*)d_in[11];

  char* wsp = (char*)d_ws;
  size_t off = 0;
  auto alloc = [&](size_t bytes) -> char* {
    char* p = wsp + off;
    off += (bytes + 255) & ~(size_t)255;
    return p;
  };
  int*            idx3  = (int*)           alloc((size_t)B_ * N1_ * 3 * 4);
  float*          w3f   = (float*)         alloc((size_t)B_ * N1_ * 3 * 4);
  unsigned short* Xhi   = (unsigned short*)alloc((size_t)B_ * CIN_ * N1_ * 2);
  unsigned short* Xlo   = (unsigned short*)alloc((size_t)B_ * CIN_ * N1_ * 2);
  unsigned short* W0hi  = (unsigned short*)alloc((size_t)M0_ * CIN_ * 2);
  unsigned short* W0lo  = (unsigned short*)alloc((size_t)M0_ * CIN_ * 2);
  unsigned short* W1hi  = (unsigned short*)alloc((size_t)M1_ * M0_ * 2);
  unsigned short* W1lo  = (unsigned short*)alloc((size_t)M1_ * M0_ * 2);
  float*          Y     = (float*)         alloc((size_t)B_ * M0_ * N1_ * 4);  // reused for layer-1 output
  float*          sc0   = (float*)         alloc(M0_ * 4);
  float*          sh0   = (float*)         alloc(M0_ * 4);
  float*          sc1   = (float*)         alloc(M1_ * 4);
  float*          sh1   = (float*)         alloc(M1_ * 4);
  // H1 reuses X's region (X is dead after GEMM0; stream-ordered)
  unsigned short* H1hi = Xhi;
  unsigned short* H1lo = Xlo;

  split_f32_kernel<<<dim3((M0_ * CIN_ + 255) / 256), 256, 0, stream>>>(w0, W0hi, W0lo, M0_ * CIN_);
  split_f32_kernel<<<dim3((M1_ * M0_ + 255) / 256), 256, 0, stream>>>(w1, W1hi, W1lo, M1_ * M0_);

  knn3_kernel<<<dim3(N1_ / 256, B_), 256, 0, stream>>>(xyz1, xyz2, idx3, w3f);
  interp_concat_kernel<<<dim3(N1_ / 128, B_), 256, 0, stream>>>(feat1, feat2, idx3, w3f, Xhi, Xlo);

  // layer 0: Y = W0 @ X   (256 x 384) @ (384 x 65536)
  wmma_gemm_kernel<<<dim3(B_ * N1_ / TBN, M0_ / TBM), 256, 0, stream>>>(W0hi, W0lo, Xhi, Xlo, Y, M0_, CIN_);
  bn_stats_kernel<<<dim3(M0_), 256, 0, stream>>>(Y, g0, be0, sc0, sh0, M0_);
  bn_relu_split_kernel<<<dim3((unsigned)((size_t)B_ * M0_ * N1_ / 256)), 256, 0, stream>>>(Y, sc0, sh0, H1hi, H1lo, M0_ - 1);

  // layer 1: Y = W1 @ H1  (128 x 256) @ (256 x 65536)
  wmma_gemm_kernel<<<dim3(B_ * N1_ / TBN, M1_ / TBM), 256, 0, stream>>>(W1hi, W1lo, H1hi, H1lo, Y, M1_, M0_);
  bn_stats_kernel<<<dim3(M1_), 256, 0, stream>>>(Y, g1, be1, sc1, sh1, M1_);
  bn_relu_out_kernel<<<dim3((unsigned)((size_t)B_ * M1_ * N1_ / 256)), 256, 0, stream>>>(Y, sc1, sh1, (float*)d_out, M1_ - 1);
}
